// Policy_Module_86053964742747
// MI455X (gfx1250) — matrix operand order repaired and checked by reading, not compiled, not run on hardware
//
#include <hip/hip_runtime.h>

// ---------------------------------------------------------------------------
// GNN forward (GENConv softmax-aggr + dense block + SAGEConv) for MI455X.
// Dense GEMMs use v_wmma_f32_16x16x32_bf16 (bf16 in, f32 accumulate).
// Edge passes use f32/i32 global atomics (L2-resident working set).
// ---------------------------------------------------------------------------

typedef __attribute__((ext_vector_type(16))) __bf16 v16bf;
typedef __attribute__((ext_vector_type(8)))  float  v8f;

#define FIN   64
#define HID   64
#define EXPD  128
#define DIM2  128
#define EPS_MSG 1e-7f
#define BN_EPS  1e-5f

#define CDIV(a, b) (((a) + (b) - 1) / (b))

// ---------------------------------------------------------------- utilities
__global__ void zero_f32_kernel(float* __restrict__ p, long n) {
  long t = (long)blockIdx.x * blockDim.x + threadIdx.x;
  if (t < n) p[t] = 0.0f;
}

__global__ void f32_to_bf16_kernel(const float* __restrict__ in,
                                   __bf16* __restrict__ out, long n) {
  long t = (long)blockIdx.x * blockDim.x + threadIdx.x;
  if (t < n) out[t] = (__bf16)in[t];
}

__device__ __forceinline__ float sigmoidf_(float v) {
  return 1.0f / (1.0f + __expf(-v));
}

// ------------------------------------------------- edge pass 1: segment max
// msg = relu(x[src]) + eps  (always > 0 -> int-compare atomicMax is valid)
__global__ void edge_max_kernel(const float* __restrict__ x,
                                const int* __restrict__ src,
                                const int* __restrict__ dst,
                                float* __restrict__ mmax, long nwork) {
  long t = (long)blockIdx.x * blockDim.x + threadIdx.x;
  if (t >= nwork) return;
  int e = (int)(t >> 4);
  int f = ((int)t & 15) * 4;
  int s = src[e], d = dst[e];
  const float4 xv = *(const float4*)(x + (size_t)s * FIN + f);
  int* mp = (int*)(mmax + (size_t)d * FIN + f);
  atomicMax(mp + 0, __float_as_int(fmaxf(xv.x, 0.0f) + EPS_MSG));
  atomicMax(mp + 1, __float_as_int(fmaxf(xv.y, 0.0f) + EPS_MSG));
  atomicMax(mp + 2, __float_as_int(fmaxf(xv.z, 0.0f) + EPS_MSG));
  atomicMax(mp + 3, __float_as_int(fmaxf(xv.w, 0.0f) + EPS_MSG));
}

// -------------------------------------- edge pass 2: softmax numer / denom
__global__ void edge_softmax_kernel(const float* __restrict__ x,
                                    const int* __restrict__ src,
                                    const int* __restrict__ dst,
                                    const float* __restrict__ mmax,
                                    float* __restrict__ denom,
                                    float* __restrict__ numer, long nwork) {
  long t = (long)blockIdx.x * blockDim.x + threadIdx.x;
  if (t >= nwork) return;
  int e = (int)(t >> 4);
  int f = ((int)t & 15) * 4;
  int s = src[e], d = dst[e];
  const float4 xv = *(const float4*)(x + (size_t)s * FIN + f);
  const float4 mv = *(const float4*)(mmax + (size_t)d * FIN + f);
  float m0 = fmaxf(xv.x, 0.0f) + EPS_MSG;
  float m1 = fmaxf(xv.y, 0.0f) + EPS_MSG;
  float m2 = fmaxf(xv.z, 0.0f) + EPS_MSG;
  float m3 = fmaxf(xv.w, 0.0f) + EPS_MSG;
  float e0 = __expf(m0 - mv.x), e1 = __expf(m1 - mv.y);
  float e2 = __expf(m2 - mv.z), e3 = __expf(m3 - mv.w);
  float* dp = denom + (size_t)d * FIN + f;
  float* np = numer + (size_t)d * FIN + f;
  atomicAdd(dp + 0, e0); atomicAdd(dp + 1, e1);
  atomicAdd(dp + 2, e2); atomicAdd(dp + 3, e3);
  atomicAdd(np + 0, e0 * m0); atomicAdd(np + 1, e1 * m1);
  atomicAdd(np + 2, e2 * m2); atomicAdd(np + 3, e3 * m3);
}

// ----------------------------- agg = numer/denom; out = agg + x  (-> bf16)
__global__ void agg_residual_kernel(const float* __restrict__ x,
                                    const float* __restrict__ denom,
                                    const float* __restrict__ numer,
                                    __bf16* __restrict__ out_bf, long n) {
  long t = (long)blockIdx.x * blockDim.x + threadIdx.x;
  if (t >= n) return;
  float den = denom[t];
  float agg = numer[t] / (den > 0.0f ? den : 1.0f);
  out_bf[t] = (__bf16)(agg + x[t]);
}

// ------------------------------------------------------------- WMMA GEMM
// C[M,Nout] (+= if accum) A[M,K](bf16,row-major) * B[K,Nout](bf16,row-major)
// act: 0 = none, 1 = relu, 2 = sigmoid.  One wave -> one 16x16 tile.
// M multiple of 16, K multiple of 32, Nout multiple of 16.
__global__ void gemm_bf16_kernel(const __bf16* __restrict__ A,
                                 const __bf16* __restrict__ B,
                                 const float* __restrict__ bias,
                                 float* __restrict__ C,
                                 int M, int K, int Nout, int act, int accum) {
  const int lane = threadIdx.x & 31;
  const int wave = threadIdx.x >> 5;
  const int nTiles = Nout >> 4;
  const int totalTiles = (M >> 4) * nTiles;
  const int tile = blockIdx.x * (blockDim.x >> 5) + wave;
  if (tile >= totalTiles) return;           // wave-uniform branch
  const int mT = tile / nTiles;
  const int nT = tile - mT * nTiles;
  const int m16 = lane & 15;                // A row (frag) / D column
  const int half = lane >> 4;

  const __bf16* Ar = A + (size_t)(mT * 16 + m16) * K;
  const __bf16* Bc = B + (size_t)nT * 16 + m16;

  v8f acc = {};
  for (int kk = 0; kk < K; kk += 32) {
    v16bf a, b;
    // A frag: lane (m16, half); elems 0..7 -> K = kk+8*half+e,
    //         elems 8..15 -> K = kk+16+8*half+(e-8)   (two 16B chunks)
    *(uint4*)&a       = *(const uint4*)(Ar + kk + 8 * half);
    *((uint4*)&a + 1) = *(const uint4*)(Ar + kk + 16 + 8 * half);
    // B frag: lane = N column (m16); elems follow A's K order (strided rows)
    const __bf16* Bp = Bc + (size_t)kk * Nout;
#pragma unroll
    for (int e = 0; e < 8; ++e) {
      b[e]     = Bp[(size_t)(8 * half + e) * Nout];
      b[8 + e] = Bp[(size_t)(16 + 8 * half + e) * Nout];
    }
    acc = __builtin_amdgcn_wmma_f32_16x16x32_bf16(
        false, a, false, b, (short)0, acc, false, false);
  }

  const int n = nT * 16 + m16;
  const float bv = bias ? bias[n] : 0.0f;
#pragma unroll
  for (int r = 0; r < 8; ++r) {
    const int row = mT * 16 + r + 8 * half;   // D: M = r + 8*half
    size_t idx = (size_t)row * Nout + n;
    float v = acc[r] + bv;
    if (accum) v += C[idx];
    if (act == 1) v = fmaxf(v, 0.0f);
    else if (act == 2) v = sigmoidf_(v);
    C[idx] = v;
  }
}

// ------------------------------------------------ batchnorm (train stats)
__global__ void bn_stats_kernel(const float* __restrict__ h,
                                float* __restrict__ sums, int nrows) {
  const int c = threadIdx.x;                // 0..127
  float s = 0.0f, sq = 0.0f;
  for (int r = blockIdx.x; r < nrows; r += gridDim.x) {
    float v = h[(size_t)r * EXPD + c];
    s += v; sq += v * v;
  }
  atomicAdd(&sums[c], s);
  atomicAdd(&sums[EXPD + c], sq);
}

__global__ void bn_finalize_kernel(const float* __restrict__ sums,
                                   const float* __restrict__ gamma,
                                   const float* __restrict__ beta,
                                   float* __restrict__ ss, int nrows) {
  int c = threadIdx.x;
  if (c >= EXPD) return;
  float inv_n = 1.0f / (float)nrows;
  float mu  = sums[c] * inv_n;
  float var = sums[EXPD + c] * inv_n - mu * mu;   // biased var
  float rs  = rsqrtf(var + BN_EPS);
  float sc  = gamma[c] * rs;
  ss[c] = sc;
  ss[EXPD + c] = beta[c] - mu * sc;
}

__global__ void bn_apply_relu_kernel(const float* __restrict__ h,
                                     const float* __restrict__ ss,
                                     __bf16* __restrict__ hn, long n) {
  long t = (long)blockIdx.x * blockDim.x + threadIdx.x;
  if (t >= n) return;
  int c = (int)t & (EXPD - 1);
  float v = h[t] * ss[c] + ss[EXPD + c];
  hn[t] = (__bf16)fmaxf(v, 0.0f);
}

// ------------------------------------- x2 = sigmoid(concat(x, h2)) -> bf16
__global__ void concat_sigmoid_kernel(const float* __restrict__ x,
                                      const float* __restrict__ h2,
                                      __bf16* __restrict__ x2bf, long n) {
  long t = (long)blockIdx.x * blockDim.x + threadIdx.x;
  if (t >= n) return;
  long row = t >> 7;
  int c = (int)t & 127;
  float v = (c < FIN) ? x[row * FIN + c] : h2[row * HID + (c - FIN)];
  x2bf[t] = (__bf16)sigmoidf_(v);
}

// ----------------------------------------- edge pass 3: SAGE sum scatter
__global__ void edge_sum_kernel(const float* __restrict__ xp,
                                const int* __restrict__ src,
                                const int* __restrict__ dst,
                                float* __restrict__ aggr, long nwork) {
  long t = (long)blockIdx.x * blockDim.x + threadIdx.x;
  if (t >= nwork) return;
  int e = (int)(t >> 5);
  int f = ((int)t & 31) * 4;
  int s = src[e], d = dst[e];
  const float4 v = *(const float4*)(xp + (size_t)s * DIM2 + f);
  float* ap = aggr + (size_t)d * DIM2 + f;
  atomicAdd(ap + 0, v.x); atomicAdd(ap + 1, v.y);
  atomicAdd(ap + 2, v.z); atomicAdd(ap + 3, v.w);
}

// --------------------------------------- logits = out2 . Wf + bf ; probs
__global__ void final_kernel(const float* __restrict__ out2,
                             const float* __restrict__ Wf,
                             const float* __restrict__ bfc,
                             float* __restrict__ dout, int n) {
  long i = (long)blockIdx.x * blockDim.x + threadIdx.x;
  if (i >= n) return;
  float s = bfc[0];
  const float* r = out2 + (size_t)i * HID;
#pragma unroll
  for (int j = 0; j < HID; ++j) s += r[j] * Wf[j];
  dout[i] = sigmoidf_(s);   // probs
  dout[n + i] = s;          // logits
}

// ===========================================================================
extern "C" void kernel_launch(void* const* d_in, const int* in_sizes, int n_in,
                              void* d_out, int out_size, void* d_ws, size_t ws_size,
                              hipStream_t stream) {
  const float* x   = (const float*)d_in[0];
  const int*   ei  = (const int*)d_in[1];
  const float* W1  = (const float*)d_in[2];
  const float* b1  = (const float*)d_in[3];
  const float* gam = (const float*)d_in[4];
  const float* bet = (const float*)d_in[5];
  const float* W2  = (const float*)d_in[6];
  const float* b2  = (const float*)d_in[7];
  const float* Wp  = (const float*)d_in[8];
  const float* bp  = (const float*)d_in[9];
  const float* Wl  = (const float*)d_in[10];
  const float* bl  = (const float*)d_in[11];
  const float* Wr  = (const float*)d_in[12];
  const float* Wf  = (const float*)d_in[13];
  const float* bfc = (const float*)d_in[14];

  const int N = in_sizes[0] / FIN;       // 50000
  const int E = in_sizes[1] / 2;         // 800000
  const int* src = ei;
  const int* dst = ei + E;
  const size_t Mpad = ((size_t)N + 15) & ~(size_t)15;   // GEMM row padding

  // ----------------- workspace layout (aliased; ~122 MB) -----------------
  char* ws = (char*)d_ws;
  size_t off = 0;
  auto take = [&](size_t bytes) {
    size_t o = off; off += (bytes + 255) & ~(size_t)255; return o;
  };
  const size_t szGEN = 3 * Mpad * FIN * 4;     // mmax|denom|numer
  const size_t szXP  = Mpad * DIM2 * 4;        // xp
  size_t o_bufA = take(szGEN > szXP ? szGEN : szXP);
  size_t o_bufH = take(Mpad * EXPD * 4);       // h, later aggr
  size_t o_bufC = take(Mpad * HID * 4);        // h2, later out2
  size_t o_obf  = take(Mpad * FIN * 2);        // GENConv mlp input (bf16)
  size_t o_hnbf = take(Mpad * EXPD * 2);       // normalized+relu (bf16)
  size_t o_x2bf = take(Mpad * DIM2 * 2);       // sigmoid concat (bf16)
  size_t o_agbf = take(Mpad * DIM2 * 2);       // aggr (bf16)
  size_t o_w1   = take((size_t)HID * EXPD * 2);
  size_t o_w2   = take((size_t)EXPD * HID * 2);
  size_t o_wp   = take((size_t)DIM2 * DIM2 * 2);
  size_t o_wl   = take((size_t)DIM2 * HID * 2);
  size_t o_wr   = take((size_t)DIM2 * HID * 2);
  size_t o_sums = take(2 * EXPD * 4);
  size_t o_ss   = take(2 * EXPD * 4);
  (void)ws_size; (void)n_in; (void)out_size;

  float*  mmax  = (float*)(ws + o_bufA);
  float*  denom = mmax + Mpad * FIN;
  float*  numer = denom + Mpad * FIN;
  float*  xp    = (float*)(ws + o_bufA);       // reuses GEN region
  float*  h     = (float*)(ws + o_bufH);
  float*  aggr  = (float*)(ws + o_bufH);       // reuses h region
  float*  h2    = (float*)(ws + o_bufC);
  float*  out2  = (float*)(ws + o_bufC);       // reuses h2 region
  __bf16* obf   = (__bf16*)(ws + o_obf);
  __bf16* hnbf  = (__bf16*)(ws + o_hnbf);
  __bf16* x2bf  = (__bf16*)(ws + o_x2bf);
  __bf16* agbf  = (__bf16*)(ws + o_agbf);
  __bf16* w1bf  = (__bf16*)(ws + o_w1);
  __bf16* w2bf  = (__bf16*)(ws + o_w2);
  __bf16* wpbf  = (__bf16*)(ws + o_wp);
  __bf16* wlbf  = (__bf16*)(ws + o_wl);
  __bf16* wrbf  = (__bf16*)(ws + o_wr);
  float*  sums  = (float*)(ws + o_sums);
  float*  ss    = (float*)(ws + o_ss);

  const int T = 256;
  auto gemm = [&](const __bf16* A, const __bf16* B, const float* bias, float* C,
                  int K, int Nout, int act, int accum) {
    int tiles = ((int)Mpad / 16) * (Nout / 16);
    gemm_bf16_kernel<<<CDIV(tiles, 8), 256, 0, stream>>>(
        A, B, bias, C, (int)Mpad, K, Nout, act, accum);
  };

  // 1) zero accumulators (GEN region + BN sums)
  {
    long n = (long)3 * Mpad * FIN;
    zero_f32_kernel<<<CDIV(n, (long)T), T, 0, stream>>>(mmax, n);
    zero_f32_kernel<<<1, 2 * EXPD, 0, stream>>>(sums, 2 * EXPD);
  }
  // 2) weights -> bf16
  f32_to_bf16_kernel<<<CDIV(HID * EXPD, T), T, 0, stream>>>(W1, w1bf, HID * EXPD);
  f32_to_bf16_kernel<<<CDIV(EXPD * HID, T), T, 0, stream>>>(W2, w2bf, EXPD * HID);
  f32_to_bf16_kernel<<<CDIV(DIM2 * DIM2, T), T, 0, stream>>>(Wp, wpbf, DIM2 * DIM2);
  f32_to_bf16_kernel<<<CDIV(DIM2 * HID, T), T, 0, stream>>>(Wl, wlbf, DIM2 * HID);
  f32_to_bf16_kernel<<<CDIV(DIM2 * HID, T), T, 0, stream>>>(Wr, wrbf, DIM2 * HID);

  // 3) GENConv softmax aggregation (two edge passes)
  {
    long nw = (long)E * 16;
    edge_max_kernel<<<CDIV(nw, (long)T), T, 0, stream>>>(x, src, dst, mmax, nw);
    edge_softmax_kernel<<<CDIV(nw, (long)T), T, 0, stream>>>(
        x, src, dst, mmax, denom, numer, nw);
  }
  // 4) agg + residual -> bf16
  {
    long n = (long)N * FIN;
    agg_residual_kernel<<<CDIV(n, (long)T), T, 0, stream>>>(x, denom, numer, obf, n);
  }
  // 5) h = out @ W1 + b1   [N,128]
  gemm(obf, w1bf, b1, h, FIN, EXPD, /*act*/0, /*accum*/0);
  // 6) batchnorm (training stats) + relu -> bf16
  bn_stats_kernel<<<512, EXPD, 0, stream>>>(h, sums, N);
  bn_finalize_kernel<<<1, EXPD, 0, stream>>>(sums, gam, bet, ss, N);
  {
    long n = (long)N * EXPD;
    bn_apply_relu_kernel<<<CDIV(n, (long)T), T, 0, stream>>>(h, ss, hnbf, n);
  }
  // 7) h2 = relu_bn @ W2 + b2   [N,64]
  gemm(hnbf, w2bf, b2, h2, EXPD, HID, 0, 0);
  // 8) x2 = sigmoid(concat(x, h2)) -> bf16
  {
    long n = (long)N * DIM2;
    concat_sigmoid_kernel<<<CDIV(n, (long)T), T, 0, stream>>>(x, h2, x2bf, n);
  }
  // 9) xp = relu(x2 @ Wp + bp)   [N,128]  (overwrites GEN region)
  gemm(x2bf, wpbf, bp, xp, DIM2, DIM2, /*relu*/1, 0);
  // 10) aggr = segment_sum(xp[src], dst)   (h region reused; zero first)
  {
    long n = (long)Mpad * DIM2;
    zero_f32_kernel<<<CDIV(n, (long)T), T, 0, stream>>>(aggr, n);
    long nw = (long)E * 32;
    edge_sum_kernel<<<CDIV(nw, (long)T), T, 0, stream>>>(xp, src, dst, aggr, nw);
    f32_to_bf16_kernel<<<CDIV(n, (long)T), T, 0, stream>>>(aggr, agbf, n);
  }
  // 11) out2 = sigmoid(aggr @ Wl + bl + x2 @ Wr)
  gemm(agbf, wlbf, bl, out2, DIM2, HID, /*none*/0, /*accum*/0);
  gemm(x2bf, wrbf, nullptr, out2, DIM2, HID, /*sigmoid*/2, /*accum*/1);
  // 12) logits / probs -> d_out  (probs [0..N), logits [N..2N))
  final_kernel<<<CDIV(N, T), T, 0, stream>>>(out2, Wf, bfc, (float*)d_out, N);
}
